// AttentionDenseBlock_out_simple_53661321396397
// MI455X (gfx1250) — matrix operand order repaired and checked by reading, not compiled, not run on hardware
//
#include <hip/hip_runtime.h>
#include <hip/hip_bf16.h>
#include <math.h>

typedef _Float16 v16h __attribute__((ext_vector_type(16)));
typedef _Float16 v8h  __attribute__((ext_vector_type(8)));
typedef _Float16 h4   __attribute__((ext_vector_type(4)));
typedef _Float16 h2   __attribute__((ext_vector_type(2)));
typedef float    v8f  __attribute__((ext_vector_type(8)));

#define BATCH 8
#define CIN   256
#define COUT  256
#define HH    56
#define WW    56
#define HWSZ  3136          // 56*56
#define KTOT  2304          // CIN*9
#define KQ    768           // K per pass = 3 taps (3 passes)
#define ROWQ  776           // KQ + 8 halfs pad -> row stride 388 dwords (4 mod 64)
#define MTILE 32            // pixels per block
#define NBLK  98            // HWSZ/MTILE pixel tiles per batch
#define NPAIR 288           // MTILE*9 (pixel,tap) pairs
#define ZSENT (-256)        // zero-page sentinel: 256 halfs right before xsn

#define ASIZE_B   (MTILE * ROWQ * 2)            // 49664 B LDS A-stage
#define WMMA_F16  __builtin_amdgcn_wmma_f32_16x16x32_f16

// async global->LDS copy, 8B per lane; IOFFSET is added to BOTH the global
// address and the LDS address (ISA 15.18.3), so one base pair serves a tap.
#define ALOAD(OFS)                                                        \
  asm volatile("global_load_async_to_lds_b64 %0, %1, off offset:" #OFS    \
               :: "v"(dsto), "v"(src) : "memory")

__device__ __forceinline__ v16h cat8(v8h lo, v8h hi) {
  return __builtin_shufflevector(lo, hi, 0, 1, 2, 3, 4, 5, 6, 7,
                                 8, 9, 10, 11, 12, 13, 14, 15);
}

// ---------------------------------------------------------------------------
// Kernel 0: weights -> [O][k] f16, k = kp*256 + c (offset weights padded 18->32)
// Also zero-fills the 256-half zero page used by border taps.
// ---------------------------------------------------------------------------
__global__ void k_prep_weights(const float* __restrict__ w_conv,
                               const float* __restrict__ w_off,
                               _Float16* __restrict__ wbc,
                               _Float16* __restrict__ wbo,
                               _Float16* __restrict__ zpage) {
  if (blockIdx.x == 0) zpage[threadIdx.x] = (_Float16)0.0f;
  int idx = blockIdx.x * 256 + threadIdx.x;
  const int NCONV = COUT * KTOT;
  if (idx < NCONV) {
    int o = idx / KTOT, r = idx - o * KTOT;
    int kp = r >> 8, c = r & 255;
    wbc[idx] = (_Float16)w_conv[(size_t)o * KTOT + c * 9 + kp];
  } else {
    int j = idx - NCONV;
    if (j < 32 * KTOT) {
      int n = j / KTOT, r = j - n * KTOT;
      int kp = r >> 8, c = r & 255;
      wbo[j] = (n < 18) ? (_Float16)w_off[(size_t)n * KTOT + c * 9 + kp]
                        : (_Float16)0.0f;
    }
  }
}

// ---------------------------------------------------------------------------
// Kernel 1: SE gate. One block per batch.
// ---------------------------------------------------------------------------
__global__ void k_se(const float* __restrict__ x,
                     const float* __restrict__ fc1,
                     const float* __restrict__ fc2,
                     float* __restrict__ sW) {
  __shared__ float yL[CIN];
  __shared__ float hL[16];
  const int b = blockIdx.x;
  const int t = threadIdx.x;
  const int lane = t & 31, wave = t >> 5;

  for (int cc = 0; cc < 32; ++cc) {
    int c = wave * 32 + cc;
    const float* px = x + ((size_t)(b * CIN + c)) * HWSZ;
    float sum = 0.f;
    for (int i = lane; i < HWSZ; i += 32) sum += px[i];
    for (int o = 16; o > 0; o >>= 1) sum += __shfl_xor(sum, o, 32);
    if (lane == 0) yL[c] = sum * (1.0f / (float)HWSZ);
  }
  __syncthreads();
  if (t < 16) {
    float a = 0.f;
    for (int c = 0; c < CIN; ++c) a += yL[c] * fc1[t * CIN + c];
    hL[t] = a > 0.f ? a : 0.f;
  }
  __syncthreads();
  {
    float a = 0.f;
    for (int r = 0; r < 16; ++r) a += hL[r] * fc2[t * 16 + r];
    sW[b * CIN + t] = 1.0f / (1.0f + expf(-a));
  }
}

// ---------------------------------------------------------------------------
// Kernel 2: xs_nhwc[b][p][c] (f16) = x[b][c][p] * s[b][c]   (LDS transpose)
// ---------------------------------------------------------------------------
__global__ void k_scale_nhwc(const float* __restrict__ x,
                             const float* __restrict__ sW,
                             _Float16* __restrict__ xsn) {
  __shared__ _Float16 tl[32][65];
  const int bi = blockIdx.x;
  const int b = bi / 392;
  const int rem = bi - 392 * b;
  const int p0 = (rem >> 3) * 64;
  const int c0 = (rem & 7) * 32;
  const int t = threadIdx.x;

  const int pl = t & 63, cq = t >> 6;
#pragma unroll
  for (int i = 0; i < 8; ++i) {
    int cl = cq * 8 + i;
    int c = c0 + cl;
    float v = x[((size_t)(b * CIN + c)) * HWSZ + p0 + pl] * sW[b * CIN + c];
    tl[cl][pl] = (_Float16)v;
  }
  __syncthreads();
  const int cl2 = (t & 15) * 2, pq = t >> 4;
#pragma unroll
  for (int j = 0; j < 4; ++j) {
    int plj = pq * 4 + j;
    h2 v;
    v[0] = tl[cl2][plj];
    v[1] = tl[cl2 + 1][plj];
    *(h2*)(xsn + ((size_t)(b * HWSZ + p0 + plj)) * CIN + c0 + cl2) = v;
  }
}

// ---------------------------------------------------------------------------
// Kernel 3: offset conv WMMA GEMM. M=32 pixels, N=32 (18 used), K=2304 in 3
// tap-aligned passes of 768. im2col fill: async global->LDS b64 copies with
// immediate offsets (1 base pair per tap). Waves 0,1 run the WMMA.
// ---------------------------------------------------------------------------
__global__ void k_offconv(const _Float16* __restrict__ xsn,
                          const _Float16* __restrict__ wbo,
                          const float* __restrict__ boff,
                          float* __restrict__ offs) {
  extern __shared__ char smem[];
  _Float16* As = (_Float16*)smem;                 // MTILE * ROWQ halfs (LDS off 0)
  int* bs = (int*)(smem + ASIZE_B);               // NPAIR ints

  const int t = threadIdx.x;
  const int b = blockIdx.x / NBLK;
  const int tilebase = (blockIdx.x - b * NBLK) * MTILE;

  // precompute (pixel,tap) neighbor element bases (zero page when OOB)
  for (int pr = t; pr < NPAIR; pr += 256) {
    int p = pr / 9, kp = pr - 9 * p;
    int ki = kp / 3, kj = kp - 3 * ki;
    int pix = tilebase + p;
    int hh = pix / WW, ww = pix - WW * hh;
    int iy = hh + ki - 1, ix = ww + kj - 1;
    bool v = (iy >= 0) && (iy < HH) && (ix >= 0) && (ix < WW);
    bs[pr] = v ? (b * HWSZ + iy * WW + ix) * CIN : ZSENT;
  }
  __syncthreads();

  const int lane = t & 31, wave = t >> 5;
  const int n = lane & 15;
  const int kbA = (lane < 16) ? 0 : 16;
  const int kbB = (lane < 16) ? 0 : 16;
  const int mbase = (lane < 16) ? 0 : 8;
  const int fp = t >> 3;          // fill: pixel row 0..31
  const int fsub = t & 7;         // fill: h4 lane within row
  const _Float16* Arow0 = As + n * ROWQ;
  const _Float16* Arow1 = As + (n + 16) * ROWQ;

  v8f acc0 = {};                  // M-tile 0
  v8f acc1 = {};                  // M-tile 1

  for (int pass = 0; pass < 3; ++pass) {
    // ---- async im2col fill: per tap, 8 b64 copies via immediate offsets ----
#pragma unroll
    for (int tap3 = 0; tap3 < 3; ++tap3) {
      int kp = pass * 3 + tap3;
      int base = bs[fp * 9 + kp];
      const _Float16* src = xsn + base + 4 * fsub;                   // +8B*fsub
      unsigned dsto = (unsigned)(fp * ROWQ + tap3 * 256 + 4 * fsub) * 2;
      ALOAD(0);   ALOAD(64);  ALOAD(128); ALOAD(192);
      ALOAD(256); ALOAD(320); ALOAD(384); ALOAD(448);
    }
    asm volatile("s_wait_asynccnt 0" ::: "memory");
    __syncthreads();
    if (wave < 2) {
      const int o = wave * 16 + n;
      const _Float16* bcol = wbo + (size_t)o * KTOT + pass * KQ + kbB;
#pragma unroll 2
      for (int kt = 0; kt < 24; ++kt) {
        int ka = kt * 32;
        v16h a0 = cat8(*(const v8h*)(Arow0 + ka + kbA),
                       *(const v8h*)(Arow0 + ka + kbA + 8));
        v16h a1 = cat8(*(const v8h*)(Arow1 + ka + kbA),
                       *(const v8h*)(Arow1 + ka + kbA + 8));
        v16h bm = *(const v16h*)(bcol + ka);
        acc0 = WMMA_F16(false, a0, false, bm, (short)0, acc0, false, false);
        acc1 = WMMA_F16(false, a1, false, bm, (short)0, acc1, false, false);
      }
    }
    __syncthreads();
  }

  if (wave < 2) {
    int ch = wave * 16 + n;
    if (ch < 18) {
      float bias = boff[ch];
#pragma unroll
      for (int e = 0; e < 8; ++e) {
        offs[((size_t)(b * HWSZ) + tilebase + mbase + e) * 18 + ch] =
            acc0[e] + bias;
        offs[((size_t)(b * HWSZ) + tilebase + 16 + mbase + e) * 18 + ch] =
            acc1[e] + bias;
      }
    }
  }
}

// ---------------------------------------------------------------------------
// Kernel 4: deformable conv. M=32 pixels, N=256, K=2304 in 3 tap-aligned
// passes of 768. Bilinear fill: unconditional h4 gathers (zero-page sentinel)
// with packed-f16 lerp. 8 waves x (2 M x 2 N) WMMA; bias + tanh on store.
// ---------------------------------------------------------------------------
__global__ void k_deform(const _Float16* __restrict__ xsn,
                         const _Float16* __restrict__ wbc,
                         const float* __restrict__ offs,
                         const float* __restrict__ bconv,
                         float* __restrict__ out) {
  extern __shared__ char smem[];
  _Float16* As = (_Float16*)smem;                            // MTILE*ROWQ halfs
  int* bs = (int*)(smem + ASIZE_B);                          // 4*NPAIR ints
  float* wt = (float*)(smem + ASIZE_B + 4 * NPAIR * 4);      // 4*NPAIR floats

  const int t = threadIdx.x;
  const int b = blockIdx.x / NBLK;
  const int tilebase = (blockIdx.x - b * NBLK) * MTILE;

  for (int pr = t; pr < NPAIR; pr += 256) {
    int p = pr / 9, kp = pr - 9 * p;
    int ki = kp / 3, kj = kp - 3 * ki;
    int pix = tilebase + p;
    int hh = pix / WW, ww = pix - WW * hh;
    const float* op = offs + ((size_t)(b * HWSZ + pix)) * 18 + 2 * kp;
    float dy = op[0], dx = op[1];
    float ysf = (float)(hh - 1 + ki) + dy;
    float xsf = (float)(ww - 1 + kj) + dx;
    float y0 = floorf(ysf), x0 = floorf(xsf);
    float wy = ysf - y0, wx = xsf - x0;
    int iy = (int)y0, ix = (int)x0;
    float tw[4] = {(1.f - wy) * (1.f - wx), (1.f - wy) * wx,
                   wy * (1.f - wx), wy * wx};
#pragma unroll
    for (int tap = 0; tap < 4; ++tap) {
      int yy = iy + (tap >> 1), xx = ix + (tap & 1);
      bool v = (yy >= 0) && (yy < HH) && (xx >= 0) && (xx < WW);
      bs[tap * NPAIR + pr] = v ? (b * HWSZ + yy * WW + xx) * CIN : ZSENT;
      wt[tap * NPAIR + pr] = tw[tap];
    }
  }
  __syncthreads();

  const int lane = t & 31, wave = t >> 5;
  const int n = lane & 15;
  const int kbA = (lane < 16) ? 0 : 16;
  const int kbB = (lane < 16) ? 0 : 16;
  const int mbase = (lane < 16) ? 0 : 8;
  const int fp = t >> 3;
  const int fsub = t & 7;
  const int o0 = wave * 32 + n;
  const int o1 = wave * 32 + 16 + n;
  const _Float16* Arow0 = As + n * ROWQ;
  const _Float16* Arow1 = As + (n + 16) * ROWQ;

  v8f acc00 = {}, acc01 = {};      // M-tile 0 x {o0, o1}
  v8f acc10 = {}, acc11 = {};      // M-tile 1 x {o0, o1}

  for (int pass = 0; pass < 3; ++pass) {
    // ---- bilinear-sample slice into LDS: packed-f16 lerp, no branches ----
#pragma unroll
    for (int tap3 = 0; tap3 < 3; ++tap3) {
      int kp = pass * 3 + tap3;
      int pair = fp * 9 + kp;
      int b0 = bs[0 * NPAIR + pair], b1 = bs[1 * NPAIR + pair];
      int b2 = bs[2 * NPAIR + pair], b3 = bs[3 * NPAIR + pair];
      _Float16 w0h = (_Float16)wt[0 * NPAIR + pair];
      _Float16 w1h = (_Float16)wt[1 * NPAIR + pair];
      _Float16 w2h = (_Float16)wt[2 * NPAIR + pair];
      _Float16 w3h = (_Float16)wt[3 * NPAIR + pair];
      _Float16* dst = As + fp * ROWQ + tap3 * 256;
#pragma unroll
      for (int it = 0; it < 8; ++it) {
        int c4o = 4 * (fsub + it * 8);
        h4 v0 = *(const h4*)(xsn + b0 + c4o);
        h4 v1 = *(const h4*)(xsn + b1 + c4o);
        h4 v2 = *(const h4*)(xsn + b2 + c4o);
        h4 v3 = *(const h4*)(xsn + b3 + c4o);
        h4 r = v0 * w0h + v1 * w1h + v2 * w2h + v3 * w3h;
        *(h4*)(dst + c4o) = r;
      }
    }
    __syncthreads();
    const _Float16* bcol0 = wbc + (size_t)o0 * KTOT + pass * KQ + kbB;
    const _Float16* bcol1 = wbc + (size_t)o1 * KTOT + pass * KQ + kbB;
#pragma unroll 2
    for (int kt = 0; kt < 24; ++kt) {
      int ka = kt * 32;
      v16h a0 = cat8(*(const v8h*)(Arow0 + ka + kbA),
                     *(const v8h*)(Arow0 + ka + kbA + 8));
      v16h a1 = cat8(*(const v8h*)(Arow1 + ka + kbA),
                     *(const v8h*)(Arow1 + ka + kbA + 8));
      v16h bm0 = *(const v16h*)(bcol0 + ka);
      v16h bm1 = *(const v16h*)(bcol1 + ka);
      acc00 = WMMA_F16(false, a0, false, bm0, (short)0, acc00, false, false);
      acc01 = WMMA_F16(false, a0, false, bm1, (short)0, acc01, false, false);
      acc10 = WMMA_F16(false, a1, false, bm0, (short)0, acc10, false, false);
      acc11 = WMMA_F16(false, a1, false, bm1, (short)0, acc11, false, false);
    }
    __syncthreads();
  }

  // ---- bias + tanh, store NCHW f32 ----
  float bias0 = bconv[o0], bias1 = bconv[o1];
  float* p00 = out + ((size_t)(b * COUT + o0)) * HWSZ + tilebase + mbase;
  float* p01 = out + ((size_t)(b * COUT + o1)) * HWSZ + tilebase + mbase;
#pragma unroll
  for (int e = 0; e < 8; ++e) {
    p00[e] = tanhf(acc00[e] + bias0);
    p00[16 + e] = tanhf(acc10[e] + bias0);
    p01[e] = tanhf(acc01[e] + bias1);
    p01[16 + e] = tanhf(acc11[e] + bias1);
  }
}

// ---------------------------------------------------------------------------
extern "C" void kernel_launch(void* const* d_in, const int* in_sizes, int n_in,
                              void* d_out, int out_size, void* d_ws, size_t ws_size,
                              hipStream_t stream) {
  (void)in_sizes; (void)n_in; (void)out_size; (void)ws_size;
  const float* x      = (const float*)d_in[0];
  const float* w_off  = (const float*)d_in[1];
  const float* b_off  = (const float*)d_in[2];
  const float* w_conv = (const float*)d_in[3];
  const float* b_conv = (const float*)d_in[4];
  const float* fc1    = (const float*)d_in[5];
  const float* fc2    = (const float*)d_in[6];

  char* ws = (char*)d_ws;
  float*    sW    = (float*)(ws + 0);                     //     8 KB
  _Float16* zpage = (_Float16*)(ws + 8192);               //   512 B (zeros)
  _Float16* xsn   = (_Float16*)(ws + 8704);               // 12.8 MB
  _Float16* wbc   = (_Float16*)(ws + 12853760);           //  1.2 MB
  _Float16* wbo   = (_Float16*)(ws + 14033408);           //  144 KB
  float*    offs  = (float*)(ws + 14180864);              //  1.8 MB

  k_prep_weights<<<2592, 256, 0, stream>>>(w_conv, w_off, wbc, wbo, zpage);
  k_se<<<BATCH, 256, 0, stream>>>(x, fc1, fc2, sW);
  k_scale_nhwc<<<BATCH * 49 * 8, 256, 0, stream>>>(x, sW, xsn);

  const size_t lds_off = ASIZE_B + NPAIR * 4;                  // 50816 B
  k_offconv<<<BATCH * NBLK, 256, lds_off, stream>>>(xsn, wbo, b_off, offs);
  const size_t lds_def = ASIZE_B + 4 * NPAIR * 4 * 2;          // 58880 B
  k_deform<<<BATCH * NBLK, 256, lds_def, stream>>>(xsn, wbc, offs, b_conv,
                                                   (float*)d_out);
}
